// HeatDiffusionAttention_47811575939094
// MI455X (gfx1250) — matrix operand order repaired and checked by reading, not compiled, not run on hardware
//
#include <hip/hip_runtime.h>
#include <hip/hip_bf16.h>
#include <math.h>

typedef __attribute__((ext_vector_type(16))) __bf16 v16bf;
typedef __attribute__((ext_vector_type(8)))  __bf16 v8bf;
typedef __attribute__((ext_vector_type(8)))  float  v8f;

struct alignas(16) F4 { float x, y, z, w; };
struct alignas(16) OCol { float zg; float zd; int ty; int pad; };

#define NHEAD  4
#define DMODEL 256
#define DK     64
#define DV     64
#define NTYPES 32
#define BB     8
#define LVLEN  1024
#define LQ     31

__device__ __forceinline__ float softplus_f(float x) {
  // numerically stable softplus using HW v_exp_f32 / v_log_f32
  return fmaxf(x, 0.0f) + __logf(1.0f + __expf(-fabsf(x)));
}

// ---------------------------------------------------------------------------
// Kernel 1: per-head attention powers -> coefficient table
// Ctab[h][tv][to] = (A1, A2, A3/2, A4/6) where A1 = pad(softmax(scores)-I)
// ---------------------------------------------------------------------------
__global__ __launch_bounds__(256) void attn_table_kernel(
    const float* __restrict__ q, const float* __restrict__ k,
    const float* __restrict__ w_qs, const float* __restrict__ w_ks,
    F4* __restrict__ Ctab)
{
  const int h = blockIdx.x;
  const int tid = threadIdx.x;
  __shared__ float sQ[LQ * DK];
  __shared__ float sK[LQ * DK];
  __shared__ float sS[LQ * LQ];
  __shared__ float sA1[NTYPES * NTYPES];
  __shared__ float sA2[NTYPES * NTYPES];
  __shared__ float sA3[NTYPES * NTYPES];
  __shared__ float sA4[NTYPES * NTYPES];

  // qh = q[1:] @ w_qs.T (per head slice), kh likewise
  for (int e = tid; e < LQ * DK; e += 256) {
    const int i = e / DK, d = e % DK;
    const float* qr = q + (1 + i) * DMODEL;
    const float* kr = k + (1 + i) * DMODEL;
    const float* wq = w_qs + (h * DK + d) * DMODEL;
    const float* wk = w_ks + (h * DK + d) * DMODEL;
    float aq = 0.f, ak = 0.f;
    for (int m = 0; m < DMODEL; ++m) { aq = fmaf(qr[m], wq[m], aq); ak = fmaf(kr[m], wk[m], ak); }
    sQ[e] = aq; sK[e] = ak;
  }
  __syncthreads();
  // scores = qh @ kh.T / sqrt(64)
  for (int e = tid; e < LQ * LQ; e += 256) {
    const int i = e / LQ, j = e % LQ;
    float s = 0.f;
    for (int d = 0; d < DK; ++d) s = fmaf(sQ[i * DK + d], sK[j * DK + d], s);
    sS[e] = s * 0.125f;
  }
  for (int e = tid; e < NTYPES * NTYPES; e += 256) sA1[e] = 0.f;
  __syncthreads();
  // softmax rows, subtract I, pad row/col 0
  if (tid < LQ) {
    const int i = tid;
    float mx = -1e30f;
    for (int j = 0; j < LQ; ++j) mx = fmaxf(mx, sS[i * LQ + j]);
    float ex[LQ], sum = 0.f;
    for (int j = 0; j < LQ; ++j) { ex[j] = __expf(sS[i * LQ + j] - mx); sum += ex[j]; }
    const float inv = 1.0f / sum;
    for (int j = 0; j < LQ; ++j)
      sA1[(i + 1) * NTYPES + (j + 1)] = ex[j] * inv - (i == j ? 1.0f : 0.f);
  }
  __syncthreads();
  // matrix powers A2..A4 (32x32)
  for (int e = tid; e < NTYPES * NTYPES; e += 256) {
    const int qq = e >> 5, kk = e & 31;
    float s = 0.f;
    for (int m = 0; m < NTYPES; ++m) s = fmaf(sA1[qq * 32 + m], sA1[m * 32 + kk], s);
    sA2[e] = s;
  }
  __syncthreads();
  for (int e = tid; e < NTYPES * NTYPES; e += 256) {
    const int qq = e >> 5, kk = e & 31;
    float s = 0.f;
    for (int m = 0; m < NTYPES; ++m) s = fmaf(sA2[qq * 32 + m], sA1[m * 32 + kk], s);
    sA3[e] = s;
  }
  __syncthreads();
  for (int e = tid; e < NTYPES * NTYPES; e += 256) {
    const int qq = e >> 5, kk = e & 31;
    float s = 0.f;
    for (int m = 0; m < NTYPES; ++m) s = fmaf(sA3[qq * 32 + m], sA1[m * 32 + kk], s);
    sA4[e] = s;
  }
  __syncthreads();
  for (int e = tid; e < NTYPES * NTYPES; e += 256) {
    F4 c;
    c.x = sA1[e]; c.y = sA2[e]; c.z = sA3[e] * 0.5f; c.w = sA4[e] * (1.0f / 6.0f);
    Ctab[h * NTYPES * NTYPES + e] = c;
  }
}

// ---------------------------------------------------------------------------
// Kernel 2a: vh = v @ w_vs.T via WMMA bf16.  C[i,n] = sum_m v[i,m]*w_vs[n,m]
// stored as vh[b][h][o][d] f32.  Block: 8 waves, 128 rows x 64 cols.
// ---------------------------------------------------------------------------
__global__ __launch_bounds__(256) void vh_gemm_kernel(
    const float* __restrict__ v, const float* __restrict__ w_vs,
    float* __restrict__ vh)
{
  const int tid = threadIdx.x;
  const int wave = tid >> 5, lane = tid & 31;
  const int m = lane & 15, halfid = lane >> 4;
  const int ntile = blockIdx.x;               // 0..3  -> 64 output cols
  const int itile = blockIdx.y;               // 0..63 -> 128 rows
  const int irow = itile * 128 + wave * 16 + m;
  const float* arow = v + (size_t)irow * DMODEL;

  v8f acc[4];
#pragma unroll
  for (int dt = 0; dt < 4; ++dt) acc[dt] = (v8f){};

  for (int kc = 0; kc < DMODEL / 32; ++kc) {
    const int mbase = kc * 32;
    // A fragment: lane row = irow, contiguous k = 16*halfid + i (matches B)
    v16bf afrag;
#pragma unroll
    for (int i = 0; i < 16; ++i) {
      const int kk = (halfid << 4) + i;
      afrag[i] = (__bf16)arow[mbase + kk];
    }
#pragma unroll
    for (int dt = 0; dt < 4; ++dt) {
      const int ncol = ntile * 64 + dt * 16 + m;
      const float* brow = w_vs + (size_t)ncol * DMODEL + mbase + halfid * 16;
      v16bf bfrag;
#pragma unroll
      for (int i = 0; i < 16; ++i) bfrag[i] = (__bf16)brow[i];
      acc[dt] = __builtin_amdgcn_wmma_f32_16x16x32_bf16(
          false, afrag, false, bfrag, (short)0, acc[dt], false, false);
    }
  }
  // scatter C/D fragment: row M = r + 8*halfid, col N = m
#pragma unroll
  for (int dt = 0; dt < 4; ++dt) {
    const int n = ntile * 64 + dt * 16 + m;
    const int h = n >> 6, d = n & 63;
#pragma unroll
    for (int r = 0; r < 8; ++r) {
      const int gi = itile * 128 + wave * 16 + r + halfid * 8;
      const int gb = gi >> 10, go = gi & 1023;
      vh[(((size_t)gb * NHEAD + h) * LVLEN + go) * DV + d] = acc[dt][r];
    }
  }
}

// ---------------------------------------------------------------------------
// Kernel 2b: zg = z @ gh1_w.T, zd = z @ ghd_w.T, ty = argmax(y)
// ---------------------------------------------------------------------------
__global__ __launch_bounds__(256) void scalars_kernel(
    const float* __restrict__ z, const float* __restrict__ y,
    const float* __restrict__ gh1_w, const float* __restrict__ ghd_w,
    float* __restrict__ zg, float* __restrict__ zd, int* __restrict__ ty)
{
  const int g = blockIdx.x * 256 + threadIdx.x;   // 0..8191 = b*1024 + l
  const float* zr = z + (size_t)g * DMODEL;
  float ag[NHEAD] = {0.f, 0.f, 0.f, 0.f};
  float ad[NHEAD] = {0.f, 0.f, 0.f, 0.f};
  for (int m = 0; m < DMODEL; ++m) {
    const float zm = zr[m];
#pragma unroll
    for (int h = 0; h < NHEAD; ++h) {
      ag[h] = fmaf(zm, gh1_w[h * DMODEL + m], ag[h]);
      ad[h] = fmaf(zm, ghd_w[h * DMODEL + m], ad[h]);
    }
  }
#pragma unroll
  for (int h = 0; h < NHEAD; ++h) {
    zg[g * NHEAD + h] = ag[h];
    zd[g * NHEAD + h] = ad[h];
  }
  const float* yr = y + (size_t)g * NTYPES;
  int t = 0; float best = yr[0];
  for (int j = 1; j < NTYPES; ++j) { const float val = yr[j]; if (val > best) { best = val; t = j; } }
  ty[g] = t;
}

// ---------------------------------------------------------------------------
// Kernel 3 (main): out[b,h,v,:] = T[b,h,v,:] @ vh[b,h] + vh[b,h,v,:]
// T built on the fly in WMMA A-fragment layout; K-loop over o in chunks of 32.
// vh chunk staged TRANSPOSED in LDS (80B row stride) so B fragments are two
// conflict-free ds_load_b128 per tile.  E = exp(-softplus(x)) = sigmoid(-x)
// -> one v_exp_f32 + v_rcp_f32 instead of 2 exp + 1 log.
// ---------------------------------------------------------------------------
__global__ __launch_bounds__(256) void heat_main_kernel(
    const F4* __restrict__ Ctab, const float* __restrict__ vh,
    const float* __restrict__ zg, const float* __restrict__ zd,
    const int* __restrict__ ty,
    const float* __restrict__ gh1_b, const float* __restrict__ ghd_b,
    float* __restrict__ out)
{
  const int tid = threadIdx.x;
  const int wave = tid >> 5, lane = tid & 31;
  const int m = lane & 15, halfid = lane >> 4;
  const int vtile = blockIdx.x;   // 0..7  (128 v-rows)
  const int h = blockIdx.y;       // 0..3
  const int b = blockIdx.z;       // 0..7

  __shared__ F4     sC[NTYPES * NTYPES];     // 16 KB coefficient table
  __shared__ __bf16 sVhT[DV][40];            // transposed vh chunk: [d][o], 80B stride
  __shared__ OCol   sO[32];                  // per-o-column scalars packed 16B

  for (int e = tid; e < NTYPES * NTYPES; e += 256) sC[e] = Ctab[h * NTYPES * NTYPES + e];

  const int vrow = vtile * 128 + wave * 16 + m;
  const float zgv = zg[(b * LVLEN + vrow) * NHEAD + h] + gh1_b[h];
  const float zdv = zd[(b * LVLEN + vrow) * NHEAD + h] + ghd_b[h];
  const int tv = ty[b * LVLEN + vrow];

  const float* vhbh = vh + ((size_t)b * NHEAD + h) * LVLEN * DV;

  v8f acc[4];
#pragma unroll
  for (int dt = 0; dt < 4; ++dt) acc[dt] = (v8f){};

  for (int kc = 0; kc < LVLEN / 32; ++kc) {
    const int obase = kc * 32;
    __syncthreads();   // previous chunk fully consumed (also covers sC on kc==0)
    // stage 32x64 vh chunk, transposed, bf16 (global reads stay coalesced)
#pragma unroll
    for (int j = 0; j < 8; ++j) {
      const int idx = tid + j * 256;               // 0..2047
      const int row = idx >> 6, col = idx & 63;    // row = o-local, col = d
      sVhT[col][row] = (__bf16)vhbh[(size_t)(obase + row) * DV + col];
    }
    if (tid < 32) {
      const int oo = obase + tid;
      OCol oc;
      oc.zg = zg[(b * LVLEN + oo) * NHEAD + h];
      oc.zd = zd[(b * LVLEN + oo) * NHEAD + h];
      oc.ty = ty[b * LVLEN + oo];
      oc.pad = 0;
      sO[tid] = oc;
    }
    __syncthreads();

    // Build T-tile A fragment: element i of lane -> k = 16*halfid + i (matches B)
    v16bf afrag;
#pragma unroll
    for (int i = 0; i < 16; ++i) {
      const int kk = (halfid << 4) + i;
      const OCol oc = sO[kk];                      // one ds_load_b128
      const F4 c = sC[tv * NTYPES + oc.ty];        // one ds_load_b128
      const float G = softplus_f(zgv - oc.zg);
      // exp(-softplus(xd)) == 1/(1+exp(xd)); exp->inf => rcp->0 (correct limit)
      const float E = __builtin_amdgcn_rcpf(1.0f + __expf(zdv - oc.zd));
      const float poly = c.x + G * (c.y + G * (c.z + G * c.w));
      afrag[i] = (__bf16)(E * G * poly);
    }
    // 4 WMMA ops reuse the expensive A fragment across all 64 D_V columns
#pragma unroll
    for (int dt = 0; dt < 4; ++dt) {
      const __bf16* bp = &sVhT[dt * 16 + m][halfid * 16];
      const v8bf lo = *(const v8bf*)bp;            // ds_load_b128, conflict-free
      const v8bf hi = *(const v8bf*)(bp + 8);      // ds_load_b128, conflict-free
      const v16bf bfrag = __builtin_shufflevector(
          lo, hi, 0, 1, 2, 3, 4, 5, 6, 7, 8, 9, 10, 11, 12, 13, 14, 15);
      acc[dt] = __builtin_amdgcn_wmma_f32_16x16x32_bf16(
          false, afrag, false, bfrag, (short)0, acc[dt], false, false);
    }
  }

  // epilogue: +vh residual, store to (B, LV, NHEAD*DV)
#pragma unroll
  for (int dt = 0; dt < 4; ++dt) {
    const int d = dt * 16 + m;
#pragma unroll
    for (int r = 0; r < 8; ++r) {
      const int gv = vtile * 128 + wave * 16 + r + halfid * 8;
      const float res = acc[dt][r] + vhbh[(size_t)gv * DV + d];
      out[((size_t)(b * LVLEN + gv)) * (NHEAD * DV) + h * DV + d] = res;
    }
  }
}

// ---------------------------------------------------------------------------
extern "C" void kernel_launch(void* const* d_in, const int* in_sizes, int n_in,
                              void* d_out, int out_size, void* d_ws, size_t ws_size,
                              hipStream_t stream) {
  const float* q     = (const float*)d_in[0];
  const float* k     = (const float*)d_in[1];
  const float* v     = (const float*)d_in[2];
  const float* z     = (const float*)d_in[3];
  const float* y     = (const float*)d_in[4];
  const float* w_qs  = (const float*)d_in[5];
  const float* w_ks  = (const float*)d_in[6];
  const float* w_vs  = (const float*)d_in[7];
  const float* gh1_w = (const float*)d_in[8];
  const float* gh1_b = (const float*)d_in[9];
  const float* ghd_w = (const float*)d_in[10];
  const float* ghd_b = (const float*)d_in[11];
  float* out = (float*)d_out;

  // workspace layout (all 16B aligned)
  char* ws = (char*)d_ws;
  F4*    Ctab = (F4*)(ws);                                   //   64 KB
  float* vhw  = (float*)(ws + (64 << 10));                   //    8 MB
  float* zgw  = (float*)(ws + (64 << 10) + (8 << 20));       //  128 KB
  float* zdw  = zgw + BB * LVLEN * NHEAD;                    //  128 KB
  int*   tyw  = (int*)(zdw + BB * LVLEN * NHEAD);            //   32 KB

  attn_table_kernel<<<dim3(NHEAD), dim3(256), 0, stream>>>(q, k, w_qs, w_ks, Ctab);
  vh_gemm_kernel<<<dim3(4, 64), dim3(256), 0, stream>>>(v, w_vs, vhw);
  scalars_kernel<<<dim3(32), dim3(256), 0, stream>>>(z, y, gh1_w, ghd_w, zgw, zdw, tyw);
  heat_main_kernel<<<dim3(8, NHEAD, BB), dim3(256), 0, stream>>>(
      Ctab, vhw, zgw, zdw, tyw, gh1_b, ghd_b, out);
}
